// SwinTransformerBlock_80788334838119
// MI455X (gfx1250) — compile-verified, matrix operand order checked by reading, not run on hardware
//
#include <hip/hip_runtime.h>
#include <hip/hip_bf16.h>
#include <cmath>

typedef _Float16 f16;
typedef _Float16 v16h __attribute__((ext_vector_type(16)));
typedef _Float16 h8   __attribute__((ext_vector_type(8)));
typedef float    v8f  __attribute__((ext_vector_type(8)));

#define DIM    384
#define HEADS  12
#define HD     32
#define NW     49
#define NPAD   64
#define SHIFT  3
#define IMG    56
#define BATCH  16
#define NWIN   1024                 // 16 images * 8*8 windows
#define MROWS  (NWIN * NPAD)        // 65536 padded window rows
#define LTOK   3136
#define MTOK   (BATCH * LTOK)       // 50176 token rows
#define QKVC   (3 * DIM)            // 1152
#define MLPH   1536

// ---- gfx1250 async global->LDS path (guarded; falls back to manual staging)
#if defined(__has_builtin)
#if __has_builtin(__builtin_amdgcn_global_load_async_to_lds_b128) && \
    __has_builtin(__builtin_amdgcn_s_wait_asynccnt)
#define HAVE_ASYNC_LDS 1
#endif
#endif
#ifndef HAVE_ASYNC_LDS
#define HAVE_ASYNC_LDS 0
#endif

#if HAVE_ASYNC_LDS
typedef int v4i __attribute__((ext_vector_type(4)));
typedef __attribute__((address_space(1))) v4i g_v4i;   // global (AS1)
typedef __attribute__((address_space(3))) v4i l_v4i;   // LDS    (AS3)
#endif

// Copy 16 bytes (8 halves) from global to LDS.
__device__ inline void copy16_g2l(const f16* gsrc, f16* ldst) {
#if HAVE_ASYNC_LDS
    __builtin_amdgcn_global_load_async_to_lds_b128(
        (g_v4i*)gsrc, (l_v4i*)ldst, 0, 0);
#else
    *(h8*)ldst = *(const h8*)gsrc;
#endif
}

union Frag16 { v16h v; h8 h[2]; };

// Per-lane WMMA operand load for 16x32 f16 A/B fragments (CDNA5 layout):
// lanes 0-15 hold K 0..7 & 16..23, lanes 16-31 hold K 8..15 & 24..31.
__device__ inline v16h load_frag_row(const f16* rowbase, int lane) {
    const int koff = (lane < 16) ? 0 : 8;
    Frag16 f;
    f.h[0] = *(const h8*)(rowbase + koff);
    f.h[1] = *(const h8*)(rowbase + koff + 16);
    return f.v;
}

// ---------------------------------------------------------------- utilities
__global__ __launch_bounds__(256) void f32_to_f16(const float* __restrict__ src,
                                                  f16* __restrict__ dst, int n) {
    int i = blockIdx.x * 256 + threadIdx.x;
    if (i < n) dst[i] = (f16)src[i];
}

// ---------------------------------------------- LN1 + shift + window partition
// One wave per padded window row; rows >= 49 are zero padding.
__global__ __launch_bounds__(256) void ln_window_partition(
    const float* __restrict__ x, const float* __restrict__ gamma,
    const float* __restrict__ beta, f16* __restrict__ Xw)
{
    const int lane = threadIdx.x & 31;
    const int wid  = threadIdx.x >> 5;
    const int row  = blockIdx.x * 8 + wid;       // 0..65535
    const int w    = row >> 6;
    const int n    = row & 63;
    f16* outp = Xw + (size_t)row * DIM;
    if (n >= NW) {
        for (int j = 0; j < 12; j++) outp[lane * 12 + j] = (f16)0.0f;
        return;
    }
    const int b = w >> 6, widx = w & 63;
    const int wy = widx >> 3, wx = widx & 7;
    const int r = n / 7, c = n % 7;
    const int hh = (wy * 7 + r + SHIFT) % IMG;
    const int ww = (wx * 7 + c + SHIFT) % IMG;
    const float* src = x + ((size_t)b * LTOK + hh * IMG + ww) * DIM;

    float vals[12], s = 0.f, s2 = 0.f;
    for (int j = 0; j < 12; j++) {
        float v = src[lane * 12 + j];
        vals[j] = v; s += v; s2 += v * v;
    }
    for (int off = 16; off >= 1; off >>= 1) {
        s  += __shfl_xor(s, off);
        s2 += __shfl_xor(s2, off);
    }
    const float mean = s * (1.0f / DIM);
    const float var  = s2 * (1.0f / DIM) - mean * mean;
    const float rstd = rsqrtf(var + 1e-5f);
    for (int j = 0; j < 12; j++) {
        int cc = lane * 12 + j;
        outp[cc] = (f16)((vals[j] - mean) * rstd * gamma[cc] + beta[cc]);
    }
}

// ------------------------------------------------------- LN2 over token rows
__global__ __launch_bounds__(256) void ln_rows(
    const float* __restrict__ x, const float* __restrict__ gamma,
    const float* __restrict__ beta, f16* __restrict__ out)
{
    const int lane = threadIdx.x & 31;
    const int wid  = threadIdx.x >> 5;
    const int row  = blockIdx.x * 8 + wid;       // < 50176
    const float* src = x + (size_t)row * DIM;
    f16* outp = out + (size_t)row * DIM;

    float vals[12], s = 0.f, s2 = 0.f;
    for (int j = 0; j < 12; j++) {
        float v = src[lane * 12 + j];
        vals[j] = v; s += v; s2 += v * v;
    }
    for (int off = 16; off >= 1; off >>= 1) {
        s  += __shfl_xor(s, off);
        s2 += __shfl_xor(s2, off);
    }
    const float mean = s * (1.0f / DIM);
    const float var  = s2 * (1.0f / DIM) - mean * mean;
    const float rstd = rsqrtf(var + 1e-5f);
    for (int j = 0; j < 12; j++) {
        int cc = lane * 12 + j;
        outp[cc] = (f16)((vals[j] - mean) * rstd * gamma[cc] + beta[cc]);
    }
}

// -------------------------------------------------------------- WMMA GEMM
// D = A(MxK) * Bw(NxK)^T + bias.  Block tile 64x128, 8 waves (2M x 4N),
// each wave owns a 32x32 tile (2x2 WMMA accumulators), K staged by 32 with
// double-buffered LDS fed by async global->LDS copies (ASYNCcnt).
// EPI: 0 = store f16, 1 = GELU + store f16,
//      2 = proj: window-reverse + roll + residual -> f32,
//      3 = fc2: + residual -> f32.
template <int EPI>
__global__ __launch_bounds__(256) void wmma_gemm(
    const f16* __restrict__ A, int lda,
    const f16* __restrict__ Bw, int K,
    const float* __restrict__ bias, int ldo,
    f16* __restrict__ outH, float* __restrict__ outF,
    const float* __restrict__ resid)
{
    __shared__ f16 As[2][64][32];
    __shared__ f16 Bs[2][128][32];

    const int tid   = threadIdx.x;
    const int lane  = tid & 31;
    const int wid   = tid >> 5;
    const int waveM = wid >> 2;                  // 0..1
    const int waveN = wid & 3;                   // 0..3
    const size_t blockM = (size_t)blockIdx.y * 64;
    const size_t blockN = (size_t)blockIdx.x * 128;

    v8f acc[2][2];
    for (int i = 0; i < 2; i++)
        for (int j = 0; j < 2; j++) acc[i][j] = {};

    const int ra = tid >> 2,  ca = (tid & 3) * 8;   // A staging coords
    const int rb = tid >> 1,  cb = (tid & 1) * 16;  // B staging coords
    const f16* gA = &A[(blockM + ra) * (size_t)lda + ca];
    const f16* gB = &Bw[(blockN + rb) * (size_t)K + cb];

    // Stage 0 (3 async ops per thread per stage: 1 for A, 2 for B)
    copy16_g2l(gA,     &As[0][ra][ca]);
    copy16_g2l(gB,     &Bs[0][rb][cb]);
    copy16_g2l(gB + 8, &Bs[0][rb][cb + 8]);

    int s = 0;
    for (int k0 = 0; k0 < K; k0 += 32, s ^= 1) {
        const bool more = (k0 + 32) < K;
        if (more) {
            // Prefetch stage k0+32 into the other buffer while computing.
            copy16_g2l(gA + k0 + 32,     &As[s ^ 1][ra][ca]);
            copy16_g2l(gB + k0 + 32,     &Bs[s ^ 1][rb][cb]);
            copy16_g2l(gB + k0 + 40,     &Bs[s ^ 1][rb][cb + 8]);
#if HAVE_ASYNC_LDS
            __builtin_amdgcn_s_wait_asynccnt(3);   // stage k0 resident
#endif
        } else {
#if HAVE_ASYNC_LDS
            __builtin_amdgcn_s_wait_asynccnt(0);
#else
            __builtin_prefetch(gA + k0 + 32, 0, 1);
#endif
        }
        __syncthreads();

        v16h af[2], bf[2];
        for (int mt = 0; mt < 2; mt++)
            af[mt] = load_frag_row(&As[s][waveM * 32 + mt * 16 + (lane & 15)][0], lane);
        for (int nt = 0; nt < 2; nt++)
            bf[nt] = load_frag_row(&Bs[s][waveN * 32 + nt * 16 + (lane & 15)][0], lane);
        for (int mt = 0; mt < 2; mt++)
            for (int nt = 0; nt < 2; nt++)
                acc[mt][nt] = __builtin_amdgcn_wmma_f32_16x16x32_f16(
                    false, af[mt], false, bf[nt], (short)0, acc[mt][nt], false, false);
        __syncthreads();   // all waves done reading buf[s] before it is re-staged
    }

    const int rowHalf = (lane >> 4) * 8;
    for (int mt = 0; mt < 2; mt++) {
        for (int nt = 0; nt < 2; nt++) {
            const size_t nb = blockN + waveN * 32 + nt * 16 + (lane & 15);
            const float bn = bias[nb];
            for (int v = 0; v < 8; v++) {
                const size_t m = blockM + waveM * 32 + mt * 16 + rowHalf + v;
                float val = acc[mt][nt][v] + bn;
                if (EPI == 0) {
                    outH[m * (size_t)ldo + nb] = (f16)val;
                } else if (EPI == 1) {
                    float g = 0.5f * val * (1.0f + erff(val * 0.70710678118f));
                    outH[m * (size_t)ldo + nb] = (f16)g;
                } else if (EPI == 2) {
                    const int w = (int)(m >> 6), n = (int)(m & 63);
                    if (n < NW) {
                        const int b = w >> 6, widx = w & 63;
                        const int wy = widx >> 3, wx = widx & 7;
                        const int r = n / 7, c = n % 7;
                        const int hh = (wy * 7 + r + SHIFT) % IMG;
                        const int ww = (wx * 7 + c + SHIFT) % IMG;
                        const size_t idx = ((size_t)b * LTOK + hh * IMG + ww) * DIM + nb;
                        outF[idx] = resid[idx] + val;
                    }
                } else {
                    const size_t idx = m * (size_t)ldo + nb;
                    outF[idx] = resid[idx] + val;
                }
            }
        }
    }
}

// -------------------------------------------------------------- attention
// One block = (window, head). 4 waves x 16 query rows. S padded to 64x64.
__global__ __launch_bounds__(128) void wmma_attention(
    const f16* __restrict__ QKV, const float* __restrict__ relTable,
    const float* __restrict__ mask, f16* __restrict__ O)
{
    __shared__ float S[4][16][64];
    __shared__ f16   P[4][16][64];
    __shared__ f16   Vt[32][64];     // V transposed: [head_dim][key]

    const int h    = blockIdx.x;
    const int w    = blockIdx.y;
    const int tid  = threadIdx.x;
    const int lane = tid & 31;
    const int wid  = tid >> 5;
    const int widx = w & 63;
    const float scale = 0.17677669529663687f;   // 32^-0.5

    // Stage V^T: thread reads 16 contiguous head-dims of one key row.
    {
        const int k = tid >> 1;
        const int dbase = (tid & 1) * 16;
        const f16* vp = QKV + (size_t)(w * 64 + k) * QKVC + 2 * DIM + h * HD + dbase;
        h8 lo = *(const h8*)vp;
        h8 hi = *(const h8*)(vp + 8);
        for (int j = 0; j < 8; j++) Vt[dbase + j][k] = lo[j];
        for (int j = 0; j < 8; j++) Vt[dbase + 8 + j][k] = hi[j];
    }
    __syncthreads();

    const int m0 = wid * 16;
    const int nq = m0 + (lane & 15);
    const v16h qf = load_frag_row(QKV + (size_t)(w * 64 + nq) * QKVC + h * HD, lane);

    // S = Q K^T (K dim = head dim = 32: single WMMA per 16x16 tile)
    for (int nt = 0; nt < 4; nt++) {
        const int nk = nt * 16 + (lane & 15);
        const v16h kf = load_frag_row(
            QKV + (size_t)(w * 64 + nk) * QKVC + DIM + h * HD, lane);
        v8f acc = {};
        acc = __builtin_amdgcn_wmma_f32_16x16x32_f16(
            false, qf, false, kf, (short)0, acc, false, false);
        const int rowHalf = (lane >> 4) * 8;
        const int col = nt * 16 + (lane & 15);
        for (int v = 0; v < 8; v++) {
            const int ml = rowHalf + v;
            const int q  = m0 + ml;
            float val;
            if (q < NW && col < NW) {
                const int ri = q / 7, ci = q % 7, rj = col / 7, cj = col % 7;
                const int rel = (ri - rj + 6) * 13 + (ci - cj + 6);
                val = acc[v] * scale + relTable[rel * HEADS + h]
                    + mask[((size_t)widx * NW + q) * NW + col];
            } else {
                val = -1e30f;                    // padded keys vanish in softmax
            }
            S[wid][ml][col] = val;
        }
    }
    __syncthreads();

    // Softmax: lanes l and l^16 cooperate on one row (32 cols each).
    {
        const int row = lane & 15;
        const int cb = (lane >> 4) * 32;
        float mx = -1e30f;
        for (int j = 0; j < 32; j++) mx = fmaxf(mx, S[wid][row][cb + j]);
        mx = fmaxf(mx, __shfl_xor(mx, 16));
        float sum = 0.f;
        for (int j = 0; j < 32; j++) {
            float e = __expf(S[wid][row][cb + j] - mx);
            S[wid][row][cb + j] = e;
            sum += e;
        }
        sum += __shfl_xor(sum, 16);
        const float inv = 1.0f / sum;
        for (int j = 0; j < 32; j++)
            P[wid][row][cb + j] = (f16)(S[wid][row][cb + j] * inv);
    }
    __syncthreads();

    // O = P V : M=16 (wave strip), N=32 head dims, K=64 keys (2 steps).
    for (int nt = 0; nt < 2; nt++) {
        v8f acc = {};
        for (int kk = 0; kk < 2; kk++) {
            const v16h pf = load_frag_row(&P[wid][lane & 15][kk * 32], lane);
            const v16h vf = load_frag_row(&Vt[nt * 16 + (lane & 15)][kk * 32], lane);
            acc = __builtin_amdgcn_wmma_f32_16x16x32_f16(
                false, pf, false, vf, (short)0, acc, false, false);
        }
        const int rowHalf = (lane >> 4) * 8;
        for (int v = 0; v < 8; v++) {
            const int nqo = m0 + rowHalf + v;
            const int col = h * HD + nt * 16 + (lane & 15);
            O[(size_t)(w * 64 + nqo) * DIM + col] = (f16)acc[v];
        }
    }
}

// ---------------------------------------------------------------- launcher
extern "C" void kernel_launch(void* const* d_in, const int* in_sizes, int n_in,
                              void* d_out, int out_size, void* d_ws, size_t ws_size,
                              hipStream_t stream) {
    (void)in_sizes; (void)n_in; (void)out_size; (void)ws_size;
    const float* x         = (const float*)d_in[0];
    const float* attn_mask = (const float*)d_in[1];
    const float* gamma1    = (const float*)d_in[2];
    const float* beta1     = (const float*)d_in[3];
    const float* qkv_w     = (const float*)d_in[4];
    const float* qkv_b     = (const float*)d_in[5];
    const float* rel_tab   = (const float*)d_in[6];
    const float* proj_w    = (const float*)d_in[7];
    const float* proj_b    = (const float*)d_in[8];
    const float* gamma2    = (const float*)d_in[9];
    const float* beta2     = (const float*)d_in[10];
    const float* fc1_w     = (const float*)d_in[11];
    const float* fc1_b     = (const float*)d_in[12];
    const float* fc2_w     = (const float*)d_in[13];
    const float* fc2_b     = (const float*)d_in[14];
    float* out = (float*)d_out;

    char* ws = (char*)d_ws;
    size_t off = 0;
    auto carve = [&](size_t bytes) {
        void* p = ws + off;
        off += (bytes + 255) & ~(size_t)255;
        return p;
    };
    f16*   qkv16  = (f16*)carve((size_t)QKVC * DIM * sizeof(f16));
    f16*   proj16 = (f16*)carve((size_t)DIM * DIM * sizeof(f16));
    f16*   fc1w16 = (f16*)carve((size_t)MLPH * DIM * sizeof(f16));
    f16*   fc2w16 = (f16*)carve((size_t)DIM * MLPH * sizeof(f16));
    f16*   Xw     = (f16*)carve((size_t)MROWS * DIM * sizeof(f16));
    f16*   QKV    = (f16*)carve((size_t)MROWS * QKVC * sizeof(f16));
    f16*   Obuf   = (f16*)carve((size_t)MROWS * DIM * sizeof(f16));
    float* x1     = (float*)carve((size_t)MTOK * DIM * sizeof(float));
    f16*   xn2    = (f16*)carve((size_t)MTOK * DIM * sizeof(f16));
    f16*   hbuf   = (f16*)carve((size_t)MTOK * MLPH * sizeof(f16));

    // Weight conversion to f16
    f32_to_f16<<<(QKVC * DIM + 255) / 256, 256, 0, stream>>>(qkv_w, qkv16, QKVC * DIM);
    f32_to_f16<<<(DIM * DIM + 255) / 256, 256, 0, stream>>>(proj_w, proj16, DIM * DIM);
    f32_to_f16<<<(MLPH * DIM + 255) / 256, 256, 0, stream>>>(fc1_w, fc1w16, MLPH * DIM);
    f32_to_f16<<<(DIM * MLPH + 255) / 256, 256, 0, stream>>>(fc2_w, fc2w16, DIM * MLPH);

    // LN1 + shifted window partition (padded to 64 rows/window, f16)
    ln_window_partition<<<MROWS / 8, 256, 0, stream>>>(x, gamma1, beta1, Xw);

    // QKV projection: [65536,384] x [1152,384]^T
    wmma_gemm<0><<<dim3(QKVC / 128, MROWS / 64), 256, 0, stream>>>(
        Xw, DIM, qkv16, DIM, qkv_b, QKVC, QKV, nullptr, nullptr);

    // Windowed attention
    wmma_attention<<<dim3(HEADS, NWIN), 128, 0, stream>>>(QKV, rel_tab, attn_mask, Obuf);

    // Output projection + window reverse + roll-back + residual -> x1 (f32)
    wmma_gemm<2><<<dim3(DIM / 128, MROWS / 64), 256, 0, stream>>>(
        Obuf, DIM, proj16, DIM, proj_b, DIM, nullptr, x1, x);

    // LN2 -> f16
    ln_rows<<<MTOK / 8, 256, 0, stream>>>(x1, gamma2, beta2, xn2);

    // FC1 + GELU: [50176,384] x [1536,384]^T
    wmma_gemm<1><<<dim3(MLPH / 128, MTOK / 64), 256, 0, stream>>>(
        xn2, DIM, fc1w16, DIM, fc1_b, MLPH, hbuf, nullptr, nullptr);

    // FC2 + residual -> out (f32)
    wmma_gemm<3><<<dim3(DIM / 128, MTOK / 64), 256, 0, stream>>>(
        hbuf, MLPH, fc2w16, MLPH, fc2_b, DIM, nullptr, out, x1);
}
